// LowerLinear_30906584662420
// MI455X (gfx1250) — compile-verified, matrix operand order checked by reading, not run on hardware
//
#include <hip/hip_runtime.h>
#include <stdint.h>

// Problem constants (from reference)
#define IN_F   4096
#define OUT_F  4096
#define M_TOT  4096          // 2 * 2048 rows of x

// Tiling
#define BM 128
#define BN 128
#define BK 64
#define LDST 80              // LDS row stride in f16 elements (BK + 16 pad, keeps 16B alignment)

typedef _Float16 v16h   __attribute__((ext_vector_type(16)));
typedef __fp16   v2fp16 __attribute__((ext_vector_type(2)));
typedef float    v8f    __attribute__((ext_vector_type(8)));

// Hardware pack: two f32 -> packed f16 pair in one VALU op (v_cvt_pk_rtz_f16_f32)
static __device__ __forceinline__ uint32_t pack_f16(float a, float b) {
    union { v2fp16 h; uint32_t u; } r;
    r.h = __builtin_amdgcn_cvt_pkrtz(a, b);
    return r.u;
}

__global__ __launch_bounds__(256)
void w4a32_gemm_f16wmma(const float* __restrict__ x,
                        const int*   __restrict__ wpacked,   // [G][4] bytes-as-int32
                        const float* __restrict__ wscale,    // [G]
                        const float* __restrict__ wqbias,    // [G]
                        const float* __restrict__ bias,      // [OUT_F]
                        float* __restrict__ out)
{
    __shared__ uint16_t Alds[2][BM * LDST];   // x tile, f16, [m][k], double buffered
    __shared__ uint16_t Blds[2][BN * LDST];   // dequantized W tile, f16, [n][k]

    const int tid   = threadIdx.x;
    const int lane  = tid & 31;
    const int wave  = tid >> 5;
    const int waveM = wave & 1;        // 2 waves along M (64 rows each)
    const int waveN = wave >> 1;       // 4 waves along N (32 cols each)
    const int lm    = lane & 15;       // row/col within 16x16 tile
    const int lh    = lane >> 4;       // K-half select per ISA A/B 16-bit layout

    const int nBase = blockIdx.x * BN;
    const int mBase = blockIdx.y * BM;

    // Loop-invariant staging coordinates:
    //   A chunks:  chunk i covers row (tid>>4)+16i, k-offset (tid&15)*4
    //   B groups:  group i covers row (tid>>3)+32i, k-group  (tid&7)
    const int rowA = tid >> 4;
    const int kcA  = (tid & 15) << 2;
    const int rowB = tid >> 3;
    const int kgB  = tid & 7;
    const float* xbase = x + (size_t)(mBase + rowA) * IN_F + kcA;

    // In-flight staging registers (software pipeline)
    float4 av[8];
    int4   pv[4];
    float  sv[4], qv[4];

    auto globalLoad = [&](int kb) {
        #pragma unroll
        for (int i = 0; i < 8; ++i)
            av[i] = *(const float4*)(xbase + (size_t)i * 16 * IN_F + kb);
        #pragma unroll
        for (int i = 0; i < 4; ++i) {
            const int gg = (nBase + rowB + 32 * i) * (IN_F / 8) + (kb >> 3) + kgB;
            pv[i] = *(const int4*)(wpacked + (size_t)gg * 4);
            sv[i] = wscale[gg];
            qv[i] = wqbias[gg];
        }
    };

    auto ldsStore = [&](int buf) {
        #pragma unroll
        for (int i = 0; i < 8; ++i) {
            uint2 pk;
            pk.x = pack_f16(av[i].x, av[i].y);
            pk.y = pack_f16(av[i].z, av[i].w);
            *(uint2*)(&Alds[buf][(rowA + 16 * i) * LDST + kcA]) = pk;
        }
        #pragma unroll
        for (int i = 0; i < 4; ++i) {
            const float s = sv[i], b = qv[i];
            const int4  p = pv[i];
            uint4 d;
            d.x = pack_f16((float)( p.x       & 15) * s + b, (float)((p.x >> 4) & 15) * s + b);
            d.y = pack_f16((float)( p.y       & 15) * s + b, (float)((p.y >> 4) & 15) * s + b);
            d.z = pack_f16((float)( p.z       & 15) * s + b, (float)((p.z >> 4) & 15) * s + b);
            d.w = pack_f16((float)( p.w       & 15) * s + b, (float)((p.w >> 4) & 15) * s + b);
            *(uint4*)(&Blds[buf][(rowB + 32 * i) * LDST + kgB * 8]) = d;
        }
    };

    // ---- prologue: stage 0 ----
    globalLoad(0);
    ldsStore(0);

    v8f acc[4][2];
    #pragma unroll
    for (int mi = 0; mi < 4; ++mi)
        #pragma unroll
        for (int ni = 0; ni < 2; ++ni)
            acc[mi][ni] = (v8f)(0.0f);

    __syncthreads();

    for (int kb = 0; kb < IN_F; kb += BK) {
        const int  cur  = (kb >> 6) & 1;
        const bool more = (kb + BK) < IN_F;

        // Issue next stage's global loads first: HBM/L2 latency hides behind WMMAs.
        if (more) {
            globalLoad(kb + BK);
            if (kb + 2 * BK < IN_F) {  // warm L2 two stages ahead (global_prefetch_b8)
                __builtin_prefetch(xbase + kb + 2 * BK, 0, 1);
                const int gg2 = (nBase + rowB) * (IN_F / 8) + ((kb + 2 * BK) >> 3) + kgB;
                __builtin_prefetch(wpacked + (size_t)gg2 * 4, 0, 1);
            }
        }

        // ---- issue ALL 12 fragment loads, fence the scheduler, then 16 WMMAs ----
        v16h afrag[2][4], bfrag[2][2];
        #pragma unroll
        for (int kk = 0; kk < 2; ++kk) {
            #pragma unroll
            for (int mi = 0; mi < 4; ++mi) {
                const uint16_t* pa =
                    &Alds[cur][(waveM * 64 + mi * 16 + lm) * LDST + kk * 32 + lh * 8];
                ((uint4*)&afrag[kk][mi])[0] = *(const uint4*)(pa);        // K 0..7   (this half)
                ((uint4*)&afrag[kk][mi])[1] = *(const uint4*)(pa + 16);   // K 16..23 (this half)
            }
            #pragma unroll
            for (int ni = 0; ni < 2; ++ni) {
                const uint16_t* pb =
                    &Blds[cur][(waveN * 32 + ni * 16 + lm) * LDST + kk * 32 + lh * 8];
                ((uint4*)&bfrag[kk][ni])[0] = *(const uint4*)(pb);
                ((uint4*)&bfrag[kk][ni])[1] = *(const uint4*)(pb + 16);
            }
        }
#if __has_builtin(__builtin_amdgcn_sched_barrier)
        __builtin_amdgcn_sched_barrier(0);   // keep all 12 ds_loads in flight before any WMMA
#endif
        #pragma unroll
        for (int kk = 0; kk < 2; ++kk)
            #pragma unroll
            for (int mi = 0; mi < 4; ++mi)
                #pragma unroll
                for (int ni = 0; ni < 2; ++ni)
                    acc[mi][ni] = __builtin_amdgcn_wmma_f32_16x16x32_f16(
                        false, afrag[kk][mi], false, bfrag[kk][ni],
                        (short)0, acc[mi][ni], false, false);

        // Convert + store next stage into the other LDS buffer (VALU co-executes with WMMA tail).
        if (more) ldsStore(1 - cur);
        __syncthreads();
    }

    // ---- epilogue: add bias, store fp32 C (ISA 16x16 f32 C layout) ----
    float bv[2];
    #pragma unroll
    for (int ni = 0; ni < 2; ++ni)
        bv[ni] = bias[nBase + waveN * 32 + ni * 16 + lm];

    #pragma unroll
    for (int mi = 0; mi < 4; ++mi) {
        const int rowb = mBase + waveM * 64 + mi * 16 + lh * 8;  // VGPR r -> M = r (+8 for hi lanes)
        #pragma unroll
        for (int ni = 0; ni < 2; ++ni) {
            const int col = nBase + waveN * 32 + ni * 16 + lm;
            #pragma unroll
            for (int r = 0; r < 8; ++r)
                out[(size_t)(rowb + r) * OUT_F + col] = acc[mi][ni][r] + bv[ni];
        }
    }
}

extern "C" void kernel_launch(void* const* d_in, const int* in_sizes, int n_in,
                              void* d_out, int out_size, void* d_ws, size_t ws_size,
                              hipStream_t stream) {
    const float* x      = (const float*)d_in[0];
    const int*   wpack  = (const int*)  d_in[1];
    const float* wscale = (const float*)d_in[2];
    const float* wqbias = (const float*)d_in[3];
    const float* bias   = (const float*)d_in[4];
    float*       out    = (float*)d_out;

    dim3 grid(OUT_F / BN, M_TOT / BM);   // 32 x 32 workgroups
    w4a32_gemm_f16wmma<<<grid, 256, 0, stream>>>(x, wpack, wscale, wqbias, bias, out);
}
